// RNN_37160057045175
// MI455X (gfx1250) — matrix operand order repaired and checked by reading, not compiled, not run on hardware
//
#include <hip/hip_runtime.h>
#include <hip/hip_bf16.h>
#include <math.h>

typedef __attribute__((ext_vector_type(16))) _Float16 v16h;
typedef __attribute__((ext_vector_type(8)))  _Float16 v8h;
typedef __attribute__((ext_vector_type(8)))  float    v8f;

constexpr int kH   = 512;   // hidden
constexpr int kB   = 256;   // batch
constexpr int kT   = 512;   // timesteps
constexpr int kI   = 96;    // input size (= 3 * 32)
constexpr int kNBLK = 32;   // persistent blocks; each owns 16 hidden columns
constexpr int kTHREADS = 256;
constexpr int kCH0 = kI / 32;   // 3  k-chunks, layer-0 input proj
constexpr int kCHH = kH / 32;   // 16 k-chunks, hidden-sized K

// ---------------------------------------------------------------- helpers ---
__device__ __forceinline__ v16h cat16(v8h lo, v8h hi) {
  return __builtin_shufflevector(lo, hi, 0,1,2,3,4,5,6,7,8,9,10,11,12,13,14,15);
}
// A-operand: lane holds K = {8h..8h+7} and {16+8h..16+8h+7}; p points at 8h.
__device__ __forceinline__ v16h load_a16(const _Float16* p) {
  v8h lo = *(const v8h*)(p);
  v8h hi = *(const v8h*)(p + 16);
  return cat16(lo, hi);
}
__device__ __forceinline__ v8f wmma16(v16h a, v16h b, v8f c) {
  return __builtin_amdgcn_wmma_f32_16x16x32_f16(false, a, false, b, (short)0, c,
                                                false, false);
}
__device__ __forceinline__ float sigmoidf_(float x) {
  return 1.0f / (1.0f + __expf(-x));
}

// ------------------------------------------------------------ prep kernels ---
__global__ void zero_u32_kernel(unsigned* p, int n) {
  for (int i = blockIdx.x * blockDim.x + threadIdx.x; i < n;
       i += gridDim.x * blockDim.x)
    p[i] = 0u;
}

__global__ void bias_kernel(const float* __restrict__ bih0, const float* __restrict__ bhh0,
                            const float* __restrict__ bih1, const float* __restrict__ bhh1,
                            float* __restrict__ bias) {
  int i = blockIdx.x * blockDim.x + threadIdx.x;
  if (i < 4 * kH) {
    bias[i]          = bih0[i] + bhh0[i];
    bias[4 * kH + i] = bih1[i] + bhh1[i];
  }
}

// x: [B,T,I] f32 -> xp: [T,B,I] f16 (contiguous per-step row-major)
__global__ void pack_x_kernel(const float* __restrict__ x, _Float16* __restrict__ xp, int n) {
  int i = blockIdx.x * blockDim.x + threadIdx.x;
  if (i >= n) return;
  int k = i % kI;
  int r = (i / kI) % kB;
  int t = i / (kI * kB);
  xp[i] = (_Float16)x[((size_t)r * kT + t) * kI + k];
}

// W: [4H, K] f32 -> B-operand swizzle, f16.
// out[((blk*CH + kc)*4 + g)*512 + lane*16 + j] = W[g*512 + blk*16 + lane%16][kc*32 + 8*(lane/16) + (j%8) + 16*(j/8)]
__global__ void pack_w_kernel(const float* __restrict__ W, _Float16* __restrict__ out,
                              int K, int CH, int n) {
  int i = blockIdx.x * blockDim.x + threadIdx.x;
  if (i >= n) return;
  int j    = i & 15;
  int lane = (i >> 4) & 31;
  int g    = (i >> 9) & 3;
  int rest = i >> 11;
  int kc   = rest % CH;
  int blk  = rest / CH;
  int outcol = g * kH + blk * 16 + (lane & 15);
  int k      = kc * 32 + (lane >> 4) * 8 + (j & 7) + (j >> 3) * 16;
  out[i] = (_Float16)W[(size_t)outcol * K + k];
}

// ---------------------------------------------------------- grid barrier ---
__device__ __forceinline__ void grid_barrier(unsigned* bar) {
  __syncthreads();
  if (threadIdx.x == 0) {
    __threadfence();
    unsigned old  = atomicAdd(bar, 1u);
    unsigned need = (old / (unsigned)kNBLK + 1u) * (unsigned)kNBLK;
    while (__hip_atomic_load(bar, __ATOMIC_RELAXED, __HIP_MEMORY_SCOPE_AGENT) < need)
      __builtin_amdgcn_s_sleep(1);
    __threadfence();
  }
  __syncthreads();
}

// ------------------------------------------------- persistent fused LSTM ---
__global__ void __launch_bounds__(kTHREADS, 1)
lstm_kernel(const _Float16* __restrict__ xp,
            const _Float16* __restrict__ w0i, const _Float16* __restrict__ w0h,
            const _Float16* __restrict__ w1i, const _Float16* __restrict__ w1h,
            const float* __restrict__ bias,
            _Float16* h0a, _Float16* h0b, _Float16* h1a, _Float16* h1b,
            float* c0, float* c1, float* h1_last, unsigned* bar) {
  // Weight slices for this block's 16 hidden cols, resident in LDS.
  // lds_w0: chunks 0..2 = Wih0^T, 3..18 = Whh0^T.  lds_w1: 0..15 = Wih1^T, 16..31 = Whh1^T.
  __shared__ __attribute__((aligned(32))) _Float16 lds_w0[(kCH0 + kCHH) * 2048];
  __shared__ __attribute__((aligned(32))) _Float16 lds_w1[(2 * kCHH) * 2048];

  const int tid  = threadIdx.x;
  const int blk  = blockIdx.x;       // 0..31 : hidden cols [blk*16, blk*16+16)
  const int lane = tid & 31;
  const int wave = tid >> 5;         // 0..7 : batch rows [wave*32, wave*32+32)
  const int halfk = lane >> 4;
  const int nrow  = lane & 15;
  const int colIdx = blk * 16 + nrow;

  for (int i = tid; i < kCH0 * 2048; i += kTHREADS)
    lds_w0[i] = w0i[(size_t)blk * kCH0 * 2048 + i];
  for (int i = tid; i < kCHH * 2048; i += kTHREADS)
    lds_w0[kCH0 * 2048 + i] = w0h[(size_t)blk * kCHH * 2048 + i];
  for (int i = tid; i < kCHH * 2048; i += kTHREADS)
    lds_w1[i] = w1i[(size_t)blk * kCHH * 2048 + i];
  for (int i = tid; i < kCHH * 2048; i += kTHREADS)
    lds_w1[kCHH * 2048 + i] = w1h[(size_t)blk * kCHH * 2048 + i];
  __syncthreads();

  float bs0[4], bs1[4];
  #pragma unroll
  for (int g = 0; g < 4; ++g) {
    bs0[g] = bias[g * kH + colIdx];
    bs1[g] = bias[4 * kH + g * kH + colIdx];
  }

  const int row0 = wave * 32 + nrow;   // A-tile 0 rows
  const int row1 = row0 + 16;          // A-tile 1 rows

  for (int t = 0; t < kT; ++t) {
    const _Float16* h0r = (t & 1) ? h0b : h0a;
    _Float16*       h0w = (t & 1) ? h0a : h0b;
    const _Float16* h1r = (t & 1) ? h1b : h1a;
    _Float16*       h1w = (t & 1) ? h1a : h1b;

    // ================= layer 0: gates = x_t @ Wih0^T + h0 @ Whh0^T =======
    v8f acc0[4], acc1[4];
    #pragma unroll
    for (int g = 0; g < 4; ++g) { v8f z = {}; acc0[g] = z; acc1[g] = z; }

    {
      const _Float16* xr0 = xp + ((size_t)t * kB + row0) * kI + halfk * 8;
      const _Float16* xr1 = xp + ((size_t)t * kB + row1) * kI + halfk * 8;
      for (int kc = 0; kc < kCH0; ++kc) {
        v16h a0 = load_a16(xr0 + kc * 32);
        v16h a1 = load_a16(xr1 + kc * 32);
        const _Float16* wb = &lds_w0[kc * 2048 + lane * 16];
        #pragma unroll
        for (int g = 0; g < 4; ++g) {
          v16h b = *(const v16h*)(wb + g * 512);
          acc0[g] = wmma16(a0, b, acc0[g]);
          acc1[g] = wmma16(a1, b, acc1[g]);
        }
      }
      const _Float16* hr0 = h0r + (size_t)row0 * kH + halfk * 8;
      const _Float16* hr1 = h0r + (size_t)row1 * kH + halfk * 8;
      for (int kc = 0; kc < kCHH; ++kc) {
        v16h a0 = load_a16(hr0 + kc * 32);
        v16h a1 = load_a16(hr1 + kc * 32);
        const _Float16* wb = &lds_w0[(kCH0 + kc) * 2048 + lane * 16];
        #pragma unroll
        for (int g = 0; g < 4; ++g) {
          v16h b = *(const v16h*)(wb + g * 512);
          acc0[g] = wmma16(a0, b, acc0[g]);
          acc1[g] = wmma16(a1, b, acc1[g]);
        }
      }
    }
    // cell update (pure per-lane): acc[g][r] is gate g at (row, colIdx)
    #pragma unroll
    for (int mt = 0; mt < 2; ++mt) {
      v8f* acc = mt ? acc1 : acc0;
      const int rbase = wave * 32 + mt * 16 + halfk * 8;
      #pragma unroll
      for (int r = 0; r < 8; ++r) {
        float ig = sigmoidf_(acc[0][r] + bs0[0]);
        float fg = sigmoidf_(acc[1][r] + bs0[1]);
        float gg = tanhf(acc[2][r] + bs0[2]);
        float og = sigmoidf_(acc[3][r] + bs0[3]);
        size_t idx = (size_t)(rbase + r) * kH + colIdx;
        float cn = fg * c0[idx] + ig * gg;
        c0[idx]  = cn;
        h0w[idx] = (_Float16)(og * tanhf(cn));
      }
    }
    grid_barrier(bar);

    // ================= layer 1: gates = h0(t) @ Wih1^T + h1 @ Whh1^T ======
    #pragma unroll
    for (int g = 0; g < 4; ++g) { v8f z = {}; acc0[g] = z; acc1[g] = z; }
    {
      const _Float16* gr0 = h0w + (size_t)row0 * kH + halfk * 8;
      const _Float16* gr1 = h0w + (size_t)row1 * kH + halfk * 8;
      for (int kc = 0; kc < kCHH; ++kc) {
        v16h a0 = load_a16(gr0 + kc * 32);
        v16h a1 = load_a16(gr1 + kc * 32);
        const _Float16* wb = &lds_w1[kc * 2048 + lane * 16];
        #pragma unroll
        for (int g = 0; g < 4; ++g) {
          v16h b = *(const v16h*)(wb + g * 512);
          acc0[g] = wmma16(a0, b, acc0[g]);
          acc1[g] = wmma16(a1, b, acc1[g]);
        }
      }
      const _Float16* hr0 = h1r + (size_t)row0 * kH + halfk * 8;
      const _Float16* hr1 = h1r + (size_t)row1 * kH + halfk * 8;
      for (int kc = 0; kc < kCHH; ++kc) {
        v16h a0 = load_a16(hr0 + kc * 32);
        v16h a1 = load_a16(hr1 + kc * 32);
        const _Float16* wb = &lds_w1[(kCHH + kc) * 2048 + lane * 16];
        #pragma unroll
        for (int g = 0; g < 4; ++g) {
          v16h b = *(const v16h*)(wb + g * 512);
          acc0[g] = wmma16(a0, b, acc0[g]);
          acc1[g] = wmma16(a1, b, acc1[g]);
        }
      }
    }
    #pragma unroll
    for (int mt = 0; mt < 2; ++mt) {
      v8f* acc = mt ? acc1 : acc0;
      const int rbase = wave * 32 + mt * 16 + halfk * 8;
      #pragma unroll
      for (int r = 0; r < 8; ++r) {
        float ig = sigmoidf_(acc[0][r] + bs1[0]);
        float fg = sigmoidf_(acc[1][r] + bs1[1]);
        float gg = tanhf(acc[2][r] + bs1[2]);
        float og = sigmoidf_(acc[3][r] + bs1[3]);
        size_t idx = (size_t)(rbase + r) * kH + colIdx;
        float cn = fg * c1[idx] + ig * gg;
        c1[idx]  = cn;
        float hv = og * tanhf(cn);
        h1w[idx] = (_Float16)hv;
        if (t == kT - 1) h1_last[idx] = hv;   // f32 copy for the FC head
      }
    }
    grid_barrier(bar);
  }
}

// ------------------------------------------------------------- FC head ---
__global__ void fc_kernel(const float* __restrict__ h1_last, const float* __restrict__ Wfc,
                          const float* __restrict__ bfc, float* __restrict__ out) {
  int b = blockIdx.x * blockDim.x + threadIdx.x;
  if (b >= kB) return;
  float a0 = bfc[0], a1 = bfc[1];
  const float* hr = h1_last + (size_t)b * kH;
  for (int k = 0; k < kH; ++k) {
    float hv = hr[k];
    a0 += hv * Wfc[k];
    a1 += hv * Wfc[kH + k];
  }
  out[b * 2 + 0] = a0;
  out[b * 2 + 1] = a1;
}

// --------------------------------------------------------------- launch ---
extern "C" void kernel_launch(void* const* d_in, const int* in_sizes, int n_in,
                              void* d_out, int out_size, void* d_ws, size_t ws_size,
                              hipStream_t stream) {
  (void)in_sizes; (void)n_in; (void)out_size; (void)ws_size;
  const float* x    = (const float*)d_in[0];
  const float* Wih0 = (const float*)d_in[1];
  const float* Whh0 = (const float*)d_in[2];
  const float* bih0 = (const float*)d_in[3];
  const float* bhh0 = (const float*)d_in[4];
  const float* Wih1 = (const float*)d_in[5];
  const float* Whh1 = (const float*)d_in[6];
  const float* bih1 = (const float*)d_in[7];
  const float* bhh1 = (const float*)d_in[8];
  const float* Wfc  = (const float*)d_in[9];
  const float* bfc  = (const float*)d_in[10];

  char* ws = (char*)d_ws;
  size_t off = 0;
  auto alloc = [&](size_t bytes) {
    size_t r = off; off += (bytes + 255) & ~(size_t)255; return r;
  };
  unsigned* bar   = (unsigned*)(ws + alloc(256));
  float*    bias  = (float*)   (ws + alloc((size_t)2 * 4 * kH * sizeof(float)));
  _Float16* xp    = (_Float16*)(ws + alloc((size_t)kT * kB * kI * 2));
  _Float16* w0i   = (_Float16*)(ws + alloc((size_t)4 * kH * kI * 2));
  _Float16* w0h   = (_Float16*)(ws + alloc((size_t)4 * kH * kH * 2));
  _Float16* w1i   = (_Float16*)(ws + alloc((size_t)4 * kH * kH * 2));
  _Float16* w1h   = (_Float16*)(ws + alloc((size_t)4 * kH * kH * 2));
  char*     state = ws + off;                 // contiguous state region start
  _Float16* h0a   = (_Float16*)(ws + alloc((size_t)kB * kH * 2));
  _Float16* h0b   = (_Float16*)(ws + alloc((size_t)kB * kH * 2));
  _Float16* h1a   = (_Float16*)(ws + alloc((size_t)kB * kH * 2));
  _Float16* h1b   = (_Float16*)(ws + alloc((size_t)kB * kH * 2));
  float*    c0    = (float*)   (ws + alloc((size_t)kB * kH * 4));
  float*    c1    = (float*)   (ws + alloc((size_t)kB * kH * 4));
  float*    h1l   = (float*)   (ws + alloc((size_t)kB * kH * 4));
  size_t state_u32 = ((char*)(h1l + (size_t)kB * kH) - state) / 4;

  // re-init every launch (ws is poisoned once; replays must be deterministic)
  zero_u32_kernel<<<1, 256, 0, stream>>>(bar, 64);
  zero_u32_kernel<<<1024, 256, 0, stream>>>((unsigned*)state, (int)state_u32);
  bias_kernel<<<(4 * kH + 255) / 256, 256, 0, stream>>>(bih0, bhh0, bih1, bhh1, bias);

  int nx = kT * kB * kI;
  pack_x_kernel<<<(nx + 255) / 256, 256, 0, stream>>>(x, xp, nx);
  int n0i = 4 * kH * kI, nhh = 4 * kH * kH;
  pack_w_kernel<<<(n0i + 255) / 256, 256, 0, stream>>>(Wih0, w0i, kI, kCH0, n0i);
  pack_w_kernel<<<(nhh + 255) / 256, 256, 0, stream>>>(Whh0, w0h, kH, kCHH, nhh);
  pack_w_kernel<<<(nhh + 255) / 256, 256, 0, stream>>>(Wih1, w1i, kH, kCHH, nhh);
  pack_w_kernel<<<(nhh + 255) / 256, 256, 0, stream>>>(Whh1, w1h, kH, kCHH, nhh);

  lstm_kernel<<<kNBLK, kTHREADS, 0, stream>>>(xp, w0i, w0h, w1i, w1h, bias,
                                              h0a, h0b, h1a, h1b, c0, c1, h1l, bar);
  fc_kernel<<<1, 256, 0, stream>>>(h1l, Wfc, bfc, (float*)d_out);
}
